// VQVAELayer_17214228922948
// MI455X (gfx1250) — matrix operand order repaired and checked by reading, not compiled, not run on hardware
//
#include <hip/hip_runtime.h>
#include <stdint.h>

#define DIM     256
#define KCODES  4096
#define NROWS   65536
#define NTILES  (KCODES / 16)        // 256 code tiles of 16 codes
#define TILE_BYTES (16 * DIM * 2)    // one bf16 B tile: 8192 bytes

typedef __attribute__((ext_vector_type(16))) __bf16         v16bf;
typedef __attribute__((ext_vector_type(16))) unsigned short v16u;
typedef __attribute__((ext_vector_type(8)))  float          v8f;

// pointer element type required by the async-load builtin: int __vector(4)
typedef int v4i __attribute__((__vector_size__(16)));
typedef __attribute__((address_space(1))) v4i* gv4i_p;   // global int4*
typedef __attribute__((address_space(3))) v4i* lv4i_p;   // LDS int4*

union BF16x16 { v16u u; v16bf b; };

static __device__ __forceinline__ unsigned short f2bf_rne(float f) {
  union { float f; unsigned u; } v; v.f = f;
  unsigned r = v.u + 0x7fffu + ((v.u >> 16) & 1u);   // round-to-nearest-even
  return (unsigned short)(r >> 16);
}

// ---- CDNA5 async global->LDS copy (16B per call), guarded fallback ----
static __device__ __forceinline__ void async_tile16(const void* g, void* l) {
#if __has_builtin(__builtin_amdgcn_global_load_async_to_lds_b128)
  __builtin_amdgcn_global_load_async_to_lds_b128(
      (gv4i_p)(uintptr_t)g,
      (lv4i_p)(unsigned)(uintptr_t)l,
      0, 0);
#else
  *(uint4*)l = *(const uint4*)g;
#endif
}

template <int N>
static __device__ __forceinline__ void wait_async() {
#if __has_builtin(__builtin_amdgcn_s_wait_asynccnt)
  __builtin_amdgcn_s_wait_asynccnt(N);
#else
  asm volatile("s_wait_asynccnt %0" :: "i"(N));
#endif
}

// ---- prepass 1: per-code squared norms + w^T (for coalesced gather) ----
__global__ void prep_wss_wT(const float* __restrict__ w,
                            float* __restrict__ wss,
                            float* __restrict__ wT) {
  int k = blockIdx.x * blockDim.x + threadIdx.x;   // 4096 threads
  float s = 0.f;
  for (int d = 0; d < DIM; ++d) {
    float v = w[(size_t)d * KCODES + k];
    s += v * v;
    wT[(size_t)k * DIM + d] = v;
  }
  wss[k] = s;
}

// ---- prepass 2: pack w into the WMMA B-operand bf16 layout ----
// B tile c (16 codes), k-chunk q (32 dims). Lane l: N = l&15,
// K = 32q + (l>>4)*8 + (i&7) + (i>>3)*16  (i = 0..15). 16 bf16 (32B) contiguous per lane.
__global__ void prep_pack(const float* __restrict__ w,
                          unsigned short* __restrict__ wp) {
  int id = blockIdx.x * blockDim.x + threadIdx.x;  // 65536 = 256c * 8q * 32l
  int c = id >> 8, q = (id >> 5) & 7, l = id & 31;
  int n = (c << 4) | (l & 15);
  int dbase = q * 32 + (l >> 4) * 8;
  unsigned short* dst = wp + ((size_t)id << 4);
  for (int i = 0; i < 16; ++i)
    dst[i] = f2bf_rne(w[(size_t)(dbase + (i & 7) + ((i >> 3) << 4)) * KCODES + n]);
}

static __device__ __forceinline__ void argmin16(float& m, int& i) {
#pragma unroll
  for (int off = 1; off < 16; off <<= 1) {
    float om = __shfl_xor(m, off, 32);
    int   oi = __shfl_xor(i, off, 32);
    if (om < m || (om == m && oi < i)) { m = om; i = oi; }
  }
}

// ---- main kernel: bf16 WMMA distances + running argmin ----
__global__ __launch_bounds__(256) void vq_argmin_kernel(
    const float* __restrict__ x, const unsigned short* __restrict__ wp,
    const float* __restrict__ wss, int* __restrict__ outIdx) {
  __shared__ __align__(32) unsigned char ldsbuf[2 * TILE_BYTES];
  __shared__ float swss[KCODES];

  const int tid  = threadIdx.x;
  const int lane = tid & 31;
  const int m    = lane & 15;
  const int hi   = lane >> 4;
  const int wave = tid >> 5;

  for (int i = tid; i < KCODES; i += 256) swss[i] = wss[i];

  // kick off async copy of B tile 0 (each thread: 2 x 16B of the 8KB tile)
  const unsigned char* wpb = (const unsigned char*)wp;
  async_tile16(wpb + tid * 16,        ldsbuf + tid * 16);
  async_tile16(wpb + 4096 + tid * 16, ldsbuf + 4096 + tid * 16);

  // Load A operands: 32 rows per wave = two 16x256 bf16 tiles in VGPRs.
  // 16-bit A layout: lane = row (mod 16); lanes<16 hold K {0..7,16..23}+32q,
  // lanes>=16 hold K {8..15,24..31}+32q.
  const int rowBase = blockIdx.x * 256 + wave * 32;
  const float* xr0 = x + (size_t)(rowBase + m) * DIM;
  const float* xr1 = x + (size_t)(rowBase + 16 + m) * DIM;
  v16bf A0[8], A1[8];
#pragma unroll
  for (int q = 0; q < 8; ++q) {
    int kb = q * 32 + hi * 8;
    BF16x16 a0, a1;
#pragma unroll
    for (int i = 0; i < 8; ++i) {
      a0.u[i]     = f2bf_rne(xr0[kb + i]);
      a0.u[i + 8] = f2bf_rne(xr0[kb + 16 + i]);
      a1.u[i]     = f2bf_rne(xr1[kb + i]);
      a1.u[i + 8] = f2bf_rne(xr1[kb + 16 + i]);
    }
    A0[q] = a0.b; A1[q] = a1.b;
  }

  float rm0[8], rm1[8]; int ri0[8], ri1[8];
#pragma unroll
  for (int j = 0; j < 8; ++j) { rm0[j] = rm1[j] = 3.4e38f; ri0[j] = ri1[j] = 0; }

  for (int t = 0; t < NTILES; ++t) {
    const unsigned char* curbuf = ldsbuf + (t & 1) * TILE_BYTES;
    if (t + 1 < NTILES) {   // prefetch next tile into other buffer
      const unsigned char* src = wpb + (size_t)(t + 1) * TILE_BYTES;
      unsigned char* dst = ldsbuf + ((t + 1) & 1) * TILE_BYTES;
      async_tile16(src + tid * 16,        dst + tid * 16);
      async_tile16(src + 4096 + tid * 16, dst + 4096 + tid * 16);
      wait_async<2>();      // oldest tile (t) complete for this wave
    } else {
      wait_async<0>();
    }
    __syncthreads();        // tile t visible to all waves

    v8f acc0 = {0.f,0.f,0.f,0.f,0.f,0.f,0.f,0.f};
    v8f acc1 = {0.f,0.f,0.f,0.f,0.f,0.f,0.f,0.f};
#pragma unroll
    for (int q = 0; q < 8; ++q) {
      v16bf b = ((const BF16x16*)(curbuf + q * 1024 + lane * 32))->b;
      acc0 = __builtin_amdgcn_wmma_f32_16x16x32_bf16(false, A0[q], false, b,
                                                     (short)0, acc0, false, false);
      acc1 = __builtin_amdgcn_wmma_f32_16x16x32_bf16(false, A1[q], false, b,
                                                     (short)0, acc1, false, false);
    }
    // d = ||w||^2 - 2 x.w   (||x||^2 is per-row constant; argmin unaffected)
    const int code = (t << 4) | m;
    const float wsq = swss[code];
#pragma unroll
    for (int j = 0; j < 8; ++j) {
      float d0 = fmaf(-2.f, acc0[j], wsq);
      if (d0 < rm0[j]) { rm0[j] = d0; ri0[j] = code; }
      float d1 = fmaf(-2.f, acc1[j], wsq);
      if (d1 < rm1[j]) { rm1[j] = d1; ri1[j] = code; }
    }
    __syncthreads();        // all waves done reading tile t before overwrite
  }

  // C-layout: VGPR j, lanes 0-15 -> row j, lanes 16-31 -> row j+8.
#pragma unroll
  for (int j = 0; j < 8; ++j) { argmin16(rm0[j], ri0[j]); argmin16(rm1[j], ri1[j]); }
  if (m == 0) {
    int base = rowBase + hi * 8;
#pragma unroll
    for (int j = 0; j < 8; ++j) {
      outIdx[base + j]      = ri0[j];
      outIdx[base + 16 + j] = ri1[j];
    }
  }
}

// ---- gather: out[n,:] = wT[idx[n],:]  (float4 coalesced) ----
__global__ void vq_gather(const float* __restrict__ wT,
                          const int* __restrict__ idx,
                          float* __restrict__ out) {
  int gid = blockIdx.x * blockDim.x + threadIdx.x;
  int n = gid >> 6;              // 64 float4 per row
  int d = (gid & 63) << 2;
  int k = idx[n];
  float4 v = *(const float4*)(wT + (size_t)k * DIM + d);
  *(float4*)(out + (size_t)n * DIM + d) = v;
}

extern "C" void kernel_launch(void* const* d_in, const int* in_sizes, int n_in,
                              void* d_out, int out_size, void* d_ws, size_t ws_size,
                              hipStream_t stream) {
  const float* x = (const float*)d_in[0];   // (64,1024,256) f32
  const float* w = (const float*)d_in[1];   // (256,4096) f32
  float* out = (float*)d_out;               // (64,1024,256) f32

  unsigned char* ws = (unsigned char*)d_ws;
  float*          wss = (float*)(ws);                                   // 16 KB
  float*          wT  = (float*)(ws + 16384);                           // 4 MB
  unsigned short* wp  = (unsigned short*)(ws + 16384 + (size_t)KCODES * DIM * 4); // 2 MB
  int*            idx = (int*)(ws + 16384 + (size_t)KCODES * DIM * 4
                                         + (size_t)KCODES * DIM * 2);   // 256 KB

  prep_wss_wT<<<KCODES / 256, 256, 0, stream>>>(w, wss, wT);
  prep_pack<<<65536 / 256, 256, 0, stream>>>(w, wp);
  vq_argmin_kernel<<<NROWS / 256, 256, 0, stream>>>(x, wp, wss, idx);
  vq_gather<<<(NROWS * (DIM / 4)) / 256, 256, 0, stream>>>(wT, idx, out);
}
